// AggregatedBilinear_34754875359428
// MI455X (gfx1250) — matrix operand order repaired and checked by reading, not compiled, not run on hardware
//
#include <hip/hip_runtime.h>
#include <hip/hip_bf16.h>

// ---------------------------------------------------------------------------
// Types for gfx1250 WMMA
// ---------------------------------------------------------------------------
typedef __attribute__((ext_vector_type(16))) __bf16 v16bf;
typedef __attribute__((ext_vector_type(8)))  float  v8f;
typedef __attribute__((ext_vector_type(4)))  float  v4f;
typedef __attribute__((ext_vector_type(8)))  unsigned int v8u;
typedef __attribute__((ext_vector_type(4)))  unsigned int v4u;

#define B_ROWS   65536
#define IN_DIM   512
#define OUT_DIM  512
#define INTERNAL 128
#define CARD     32
#define WIDTH    4

// fragment-ready bf16 weight layout in workspace (dwords):
//  W1bf  : frags[8 ntile][16 kstep], each frag = 32 lanes * 8 dwords = 256 dw
//  W2bf  : same, offset +32768 dw
//  Woutbf: frags[32 ntile][4 kstep], offset +65536 dw
#define W1_OFF   0
#define W2_OFF   32768
#define WO_OFF   65536
#define CONV_TOTAL 98304   // total dwords to produce

// ---------------------------------------------------------------------------
// helpers
// ---------------------------------------------------------------------------
// round-to-nearest-even fp32->bf16 pair, packed into one dword.
// merge done with a single v_perm_b32: dst = {hi[3],hi[2],lo[3],lo[2]} bytes
__device__ __forceinline__ unsigned pkbf(float lo, float hi) {
    unsigned a = __builtin_bit_cast(unsigned, lo);
    unsigned b = __builtin_bit_cast(unsigned, hi);
    a += 0x7fffu + ((a >> 16) & 1u);
    b += 0x7fffu + ((b >> 16) & 1u);
    // src0 = b (bytes 4..7), src1 = a (bytes 0..3); selector bytes (3..0) = 7,6,3,2
    return __builtin_amdgcn_perm(b, a, 0x07060302u);
}

__device__ __forceinline__ unsigned short f2bfbits(float f) {
    unsigned a = __builtin_bit_cast(unsigned, f);
    a += 0x7fffu + ((a >> 16) & 1u);
    return (unsigned short)(a >> 16);
}

__device__ __forceinline__ float eluf(float x) {
    return x > 0.0f ? x : (__expf(x) - 1.0f);
}

__device__ __forceinline__ v8f wmma_bf16(v16bf a, v16bf b, v8f c) {
    // emits v_wmma_f32_16x16x32_bf16
    return __builtin_amdgcn_wmma_f32_16x16x32_bf16(
        /*neg_a=*/false, a, /*neg_b=*/false, b,
        /*c_mod=*/(short)0, c, /*reuse_a=*/false, /*reuse_b=*/false);
}

// Load A fragment (16x32 bf16) from a row-major fp32 matrix (streamed -> NT).
// ISA layout: lane<16  -> row lane,    K = k0+0..7  and k0+16..23
//             lane>=16 -> row lane-16, K = k0+8..15 and k0+24..31
__device__ __forceinline__ v16bf load_a_f32_nt(const float* __restrict__ base,
                                               int row, int ldm, int k0, int lane) {
    int ofs = k0 + ((lane & 16) ? 8 : 0);
    const v4f* p = (const v4f*)(base + (long)row * ldm + ofs);
    v4f f0 = __builtin_nontemporal_load(p);
    v4f f1 = __builtin_nontemporal_load(p + 1);
    const v4f* q = (const v4f*)(base + (long)row * ldm + ofs + 16);
    v4f f2 = __builtin_nontemporal_load(q);
    v4f f3 = __builtin_nontemporal_load(q + 1);
    v8u u;
    u[0] = pkbf(f0[0], f0[1]); u[1] = pkbf(f0[2], f0[3]);
    u[2] = pkbf(f1[0], f1[1]); u[3] = pkbf(f1[2], f1[3]);
    u[4] = pkbf(f2[0], f2[1]); u[5] = pkbf(f2[2], f2[3]);
    u[6] = pkbf(f3[0], f3[1]); u[7] = pkbf(f3[2], f3[3]);
    return __builtin_bit_cast(v16bf, u);
}

// Load pre-formatted B fragment: 32 bytes per lane, fully coalesced, L2-hot
__device__ __forceinline__ v16bf load_b_frag(const unsigned* __restrict__ frags,
                                             int fragIdx, int lane) {
    const v8u* p = (const v8u*)(frags + (long)fragIdx * 256 + lane * 8);
    return __builtin_bit_cast(v16bf, *p);
}

// ---------------------------------------------------------------------------
// Kernel 0: convert W1/W2/Wout fp32 -> bf16 packed in B-fragment layout
// B = W^T, so B[k][n] = W[n][k]
// ---------------------------------------------------------------------------
__global__ void __launch_bounds__(256)
convert_weights_kernel(const float* __restrict__ W1,
                       const float* __restrict__ W2,
                       const float* __restrict__ Wout,
                       unsigned* __restrict__ ws) {
    int tid = blockIdx.x * 256 + threadIdx.x;
    if (tid >= CONV_TOTAL) return;
    int region = tid >> 15;          // 0:W1  1:W2  2:Wout  (32768 dwords each)
    int r      = tid & 32767;
    int frag   = r >> 8;             // fragment index
    int lane   = (r >> 3) & 31;
    int dv     = r & 7;
    int c      = lane & 15;                 // N within tile
    int khalf  = (lane >> 4) * 8;           // K half per B layout
    unsigned outw;
    if (region < 2) {
        const float* W = (region == 0) ? W1 : W2;   // [INTERNAL][IN_DIM]
        int nt = frag >> 4;           // 0..7
        int ks = frag & 15;           // 0..15
        int n  = nt * 16 + c;
        int k  = ks * 32 + khalf + 2 * dv + ((dv & 4) ? 8 : 0);
        outw = pkbf(W[n * IN_DIM + k], W[n * IN_DIM + k + 1]);
    } else {
        // Wout: [OUT_DIM][INTERNAL]
        int nt = frag >> 2;           // 0..31
        int ks = frag & 3;            // 0..3
        int n  = nt * 16 + c;
        int k  = ks * 32 + khalf + 2 * dv + ((dv & 4) ? 8 : 0);
        outw = pkbf(Wout[n * INTERNAL + k], Wout[n * INTERNAL + k + 1]);
    }
    ws[tid] = outw;
}

// ---------------------------------------------------------------------------
// Kernel 1: fused  elu(in1@W1^T), elu(in2@W2^T), grouped bilinear + elu,
//           out = y @ Wout^T
// block = 256 threads (8 wave32), each wave owns 16 batch rows,
// grid = 65536/128 = 512 blocks
// ---------------------------------------------------------------------------
__global__ void __launch_bounds__(256)
aggregated_bilinear_fused_kernel(const float* __restrict__ in1,
                                 const float* __restrict__ in2,
                                 const float* __restrict__ Wb,
                                 const unsigned* __restrict__ ws,
                                 float* __restrict__ out) {
    __shared__ float s_Wb[CARD * WIDTH * WIDTH * WIDTH];            // 8 KB
    __shared__ __align__(16) unsigned short s_y[8 * 16 * INTERNAL]; // 32 KB bf16 bits

    const unsigned* W1bf = ws + W1_OFF;
    const unsigned* W2bf = ws + W2_OFF;
    const unsigned* WObf = ws + WO_OFF;

    const int tid  = threadIdx.x;
    const int wave = tid >> 5;
    const int lane = tid & 31;
    const int c     = lane & 15;            // N-within-tile / column owner
    const int halfm = (lane >> 4) * 8;      // C-fragment M offset
    const int m0    = blockIdx.x * 128 + wave * 16;

    // stage Wb into LDS
    for (int i = tid; i < CARD * 64; i += 256) s_Wb[i] = Wb[i];
    __syncthreads();

    // ---------------- Phase 1: x1 = in1@W1^T, x2 = in2@W2^T (pre-ELU) -------
    v8f acc1[8], acc2[8];
    const v8f vzero = {0.f, 0.f, 0.f, 0.f, 0.f, 0.f, 0.f, 0.f};
#pragma unroll
    for (int t = 0; t < 8; ++t) { acc1[t] = vzero; acc2[t] = vzero; }

    const int arow = m0 + c;
#pragma unroll 1
    for (int ks = 0; ks < 16; ++ks) {
        const int k0 = ks * 32;
        v16bf a1 = load_a_f32_nt(in1, arow, IN_DIM, k0, lane);
        v16bf a2 = load_a_f32_nt(in2, arow, IN_DIM, k0, lane);
#pragma unroll
        for (int t = 0; t < 8; ++t) {
            v16bf b1 = load_b_frag(W1bf, t * 16 + ks, lane);
            acc1[t] = wmma_bf16(a1, b1, acc1[t]);
            v16bf b2 = load_b_frag(W2bf, t * 16 + ks, lane);
            acc2[t] = wmma_bf16(a2, b2, acc2[t]);
        }
    }

    // ---------------- Phase 2: ELU + grouped bilinear + ELU -> LDS(bf16) ----
    // C-fragment: lane holds (m = halfm+e, n = t*16 + c); group g = n/4,
    // output slot o = n%4 = lane&3; siblings are lanes (lane&~3)+i.
    const int basel = lane & ~3;
    const int oo    = lane & 3;
    unsigned short* myY = s_y + wave * (16 * INTERNAL);
#pragma unroll 1
    for (int t = 0; t < 8; ++t) {
        const int g = t * 4 + (c >> 2);
        float wb[16];
#pragma unroll
        for (int q = 0; q < 16; ++q) wb[q] = s_Wb[g * 16 * 4 + oo * 16 + q];
#pragma unroll
        for (int e = 0; e < 8; ++e) {
            float x1v = eluf(acc1[t][e]);
            float x2v = eluf(acc2[t][e]);
            float av0 = __shfl(x1v, basel + 0, 32);
            float av1 = __shfl(x1v, basel + 1, 32);
            float av2 = __shfl(x1v, basel + 2, 32);
            float av3 = __shfl(x1v, basel + 3, 32);
            float bv0 = __shfl(x2v, basel + 0, 32);
            float bv1 = __shfl(x2v, basel + 1, 32);
            float bv2 = __shfl(x2v, basel + 2, 32);
            float bv3 = __shfl(x2v, basel + 3, 32);
            float y = 0.0f;
            y += av0 * (wb[0]  * bv0 + wb[1]  * bv1 + wb[2]  * bv2 + wb[3]  * bv3);
            y += av1 * (wb[4]  * bv0 + wb[5]  * bv1 + wb[6]  * bv2 + wb[7]  * bv3);
            y += av2 * (wb[8]  * bv0 + wb[9]  * bv1 + wb[10] * bv2 + wb[11] * bv3);
            y += av3 * (wb[12] * bv0 + wb[13] * bv1 + wb[14] * bv2 + wb[15] * bv3);
            y = eluf(y);
            const int m = halfm + e;
            myY[m * INTERNAL + t * 16 + c] = f2bfbits(y);
        }
    }
    __syncthreads();

    // ---------------- Phase 3: out = Y @ Wout^T -----------------------------
    // Reload Y as WMMA A-fragments from LDS (two contiguous 16B segments/lane)
    v16bf ay[4];
    const int aofs = (lane & 16) ? 8 : 0;
#pragma unroll
    for (int ks = 0; ks < 4; ++ks) {
        const int ofs = ks * 32 + aofs;
        const v4u* p0 = (const v4u*)(myY + c * INTERNAL + ofs);
        const v4u* p1 = (const v4u*)(myY + c * INTERNAL + ofs + 16);
        v4u lo = *p0;
        v4u hi = *p1;
        v8u u = __builtin_shufflevector(lo, hi, 0, 1, 2, 3, 4, 5, 6, 7);
        ay[ks] = __builtin_bit_cast(v16bf, u);
    }

#pragma unroll 1
    for (int nt = 0; nt < 32; ++nt) {
        v8f acc = vzero;
#pragma unroll
        for (int ks = 0; ks < 4; ++ks) {
            v16bf b = load_b_frag(WObf, nt * 4 + ks, lane);
            acc = wmma_bf16(ay[ks], b, acc);
        }
        const int col = nt * 16 + c;
#pragma unroll
        for (int mv = 0; mv < 8; ++mv) {
            // streamed output, written once: non-temporal store
            __builtin_nontemporal_store(acc[mv],
                out + (long)(m0 + halfm + mv) * OUT_DIM + col);
        }
    }
}

// ---------------------------------------------------------------------------
// Launch
// ---------------------------------------------------------------------------
extern "C" void kernel_launch(void* const* d_in, const int* in_sizes, int n_in,
                              void* d_out, int out_size, void* d_ws, size_t ws_size,
                              hipStream_t stream) {
    const float* input1 = (const float*)d_in[0];   // [B, 512]
    const float* input2 = (const float*)d_in[1];   // [B, 512]
    const float* W1     = (const float*)d_in[2];   // [128, 512]
    const float* W2     = (const float*)d_in[3];   // [128, 512]
    const float* Wout   = (const float*)d_in[4];   // [512, 128]
    const float* Wb     = (const float*)d_in[5];   // [32, 4, 4, 4]
    float*       out    = (float*)d_out;           // [B, 512]
    unsigned*    ws     = (unsigned*)d_ws;         // >= 384 KB used

    // 1) weights -> bf16 fragment layout (tiny, one-shot)
    convert_weights_kernel<<<(CONV_TOTAL + 255) / 256, 256, 0, stream>>>(
        W1, W2, Wout, ws);

    // 2) fused forward
    aggregated_bilinear_fused_kernel<<<B_ROWS / 128, 256, 0, stream>>>(
        input1, input2, Wb, ws, out);
}
